// MultiheadAttentionBlock_84198538871086
// MI455X (gfx1250) — matrix operand order repaired and checked by reading, not compiled, not run on hardware
//
#include <hip/hip_runtime.h>

typedef __bf16 bf16;
typedef __attribute__((ext_vector_type(16))) __bf16 bf16x16;
typedef __attribute__((ext_vector_type(8)))  __bf16 bf16x8;
typedef __attribute__((ext_vector_type(8)))  float  f32x8;

#define EMBED 1024
#define HEADS 16
#define HDIM  64
#define BATCH 2
#define SEQ   2048
#define MTOT  (BATCH*SEQ)   /* 4096 rows of x */

// ---------- helpers ----------

static __device__ inline bf16 f2bf(float f) {
    unsigned u = __builtin_bit_cast(unsigned, f);
    unsigned short s = (unsigned short)((u + 0x7FFFu + ((u >> 16) & 1u)) >> 16);
    return __builtin_bit_cast(bf16, s);
}

// 16 contiguous bf16 -> B-operand fragment (caller passes p already at k0 + half*16)
static __device__ inline bf16x16 load16(const bf16* p) {
    bf16x8 lo = *reinterpret_cast<const bf16x8*>(p);
    bf16x8 hi = *reinterpret_cast<const bf16x8*>(p + 8);
    bf16x16 r;
#pragma unroll
    for (int i = 0; i < 8; ++i) { r[i] = lo[i]; r[i + 8] = hi[i]; }
    return r;
}

// A-operand fragment: p = row base + k0; element i -> k=half*16+i
static __device__ inline bf16x16 loadA(const bf16* p, int half) {
    bf16x8 lo = *reinterpret_cast<const bf16x8*>(p + half * 16);
    bf16x8 hi = *reinterpret_cast<const bf16x8*>(p + half * 16 + 8);
    bf16x16 r;
#pragma unroll
    for (int i = 0; i < 8; ++i) { r[i] = lo[i]; r[i + 8] = hi[i]; }
    return r;
}

static __device__ inline f32x8 wmma_bf16(bf16x16 a, bf16x16 b, f32x8 c) {
    return __builtin_amdgcn_wmma_f32_16x16x32_bf16(false, a, false, b, (short)0, c,
                                                   false, false);
}

// ---------- fp32 -> bf16 convert ----------

__global__ void cvt_f32_bf16(const float* __restrict__ in, bf16* __restrict__ out, int n) {
    int i = blockIdx.x * blockDim.x + threadIdx.x;
    int stride = gridDim.x * blockDim.x;
    for (; i < n; i += stride) out[i] = f2bf(in[i]);
}

// ---------- generic WMMA GEMM: C[M,N] = A[M,K] * W[N,K]^T + bias ----------
// 4-wave block, 128x128 block tile, 64x64 per wave (4x4 WMMA register blocking).
// mode 0: bf16 out, [B,H,S,Dh] split-heads layout   (Q, K projections)
// mode 1: bf16 out, [B,H,Dh,S] transposed layout    (V projection)
// mode 2: fp32 out, row-major [M,N]                 (final projection)

__global__ __launch_bounds__(128)
void gemm_bf16_wmma(const bf16* __restrict__ A, const bf16* __restrict__ W,
                    const float* __restrict__ bias, void* __restrict__ outp, int mode) {
    const int lane = threadIdx.x & 31;
    const int wave = threadIdx.x >> 5;
    const int ln   = lane & 15;
    const int half = lane >> 4;

    const int m0 = blockIdx.y * 128 + (wave >> 1) * 64;
    const int n0 = blockIdx.x * 128 + (wave & 1) * 64;

    const bf16* arow[4];
    const bf16* wrow[4];
#pragma unroll
    for (int ti = 0; ti < 4; ++ti) arow[ti] = A + (size_t)(m0 + ti * 16 + ln) * EMBED;
#pragma unroll
    for (int tj = 0; tj < 4; ++tj) wrow[tj] = W + (size_t)(n0 + tj * 16 + ln) * EMBED;

    f32x8 acc[4][4] = {};

    for (int k0 = 0; k0 < EMBED; k0 += 32) {
        bf16x16 afrag[4], bfrag[4];
#pragma unroll
        for (int ti = 0; ti < 4; ++ti) {
            __builtin_prefetch(arow[ti] + k0 + 64, 0, 1);   // speculative, unconditional
            afrag[ti] = loadA(arow[ti] + k0, half);
        }
#pragma unroll
        for (int tj = 0; tj < 4; ++tj) {
            __builtin_prefetch(wrow[tj] + k0 + 64, 0, 1);
            bfrag[tj] = load16(wrow[tj] + k0 + half * 16);
        }
#pragma unroll
        for (int ti = 0; ti < 4; ++ti)
#pragma unroll
            for (int tj = 0; tj < 4; ++tj)
                acc[ti][tj] = wmma_bf16(afrag[ti], bfrag[tj], acc[ti][tj]);
    }

#pragma unroll
    for (int tj = 0; tj < 4; ++tj) {
        const int n  = n0 + tj * 16 + ln;
        const float bv = bias[n];
#pragma unroll
        for (int ti = 0; ti < 4; ++ti) {
#pragma unroll
            for (int r = 0; r < 8; ++r) {
                const int m = m0 + ti * 16 + r + 8 * half;
                const float v = acc[ti][tj][r] + bv;
                if (mode == 2) {
                    ((float*)outp)[(size_t)m * EMBED + n] = v;
                } else {
                    const int b  = m >> 11;        // m / SEQ
                    const int si = m & (SEQ - 1);
                    const int h  = n >> 6;         // n / HDIM
                    const int d  = n & (HDIM - 1);
                    size_t idx = (mode == 0)
                        ? ((((size_t)b * HEADS + h) * SEQ + si) * HDIM + d)
                        : ((((size_t)b * HEADS + h) * HDIM + d) * SEQ + si);
                    ((bf16*)outp)[idx] = f2bf(v);
                }
            }
        }
    }
}

// ---------- flash attention: one wave per (b, h, 16-row Q tile) ----------
// Qb, Kb: [B,H,S,Dh] bf16; Vtb: [B,H,Dh,S] bf16; mask: [S,S] fp32 (broadcast dims dropped)
// attnb out: [B,S,H*Dh] bf16

__global__ __launch_bounds__(128)
void attn_wmma(const bf16* __restrict__ Qb, const bf16* __restrict__ Kb,
               const bf16* __restrict__ Vtb, const float* __restrict__ mask,
               bf16* __restrict__ attnb) {
    __shared__ bf16 Plds[4 * 16 * 32];   // per-wave 16x32 P tile

    const int lane = threadIdx.x & 31;
    const int wave = threadIdx.x >> 5;
    const int ln   = lane & 15;
    const int half = lane >> 4;

    const int w  = blockIdx.x * 4 + wave;          // 0 .. B*H*(S/16)-1
    const int qt = w & 127;                        // S/16 = 128 tiles
    const int h  = (w >> 7) & (HEADS - 1);
    const int b  = w >> 11;
    const int q0 = qt * 16;

    const bf16* Qbase  = Qb  + (((size_t)b * HEADS + h) * SEQ) * HDIM;
    const bf16* Kbase  = Kb  + (((size_t)b * HEADS + h) * SEQ) * HDIM;
    const bf16* Vtbase = Vtb + (((size_t)b * HEADS + h) * HDIM) * SEQ;

    // Q tile A-fragments (16 x 64, two k-chunks of 32), loaded once
    const bf16* qrow = Qbase + (size_t)(q0 + ln) * HDIM;
    bf16x16 aq0 = loadA(qrow, half);
    bf16x16 aq1 = loadA(qrow + 32, half);

    f32x8 o[4] = {};
    float m_r[8], l_r[8];
#pragma unroll
    for (int r = 0; r < 8; ++r) { m_r[r] = -1.0e30f; l_r[r] = 0.0f; }

    for (int t0 = 0; t0 < SEQ; t0 += 32) {
        // ---- scores: two 16x16 tiles (t cols t0..t0+15, t0+16..t0+31) ----
        f32x8 s[2];
#pragma unroll
        for (int tt = 0; tt < 2; ++tt) {
            const bf16* krow = Kbase + (size_t)(t0 + tt * 16 + ln) * HDIM;
            bf16x16 bk0 = load16(krow + half * 16);        // k = 0..31
            bf16x16 bk1 = load16(krow + 32 + half * 16);   // k = 32..63
            f32x8 c = {};
            c = wmma_bf16(aq0, bk0, c);
            c = wmma_bf16(aq1, bk1, c);
#pragma unroll
            for (int r = 0; r < 8; ++r) {
                const int q = q0 + r + 8 * half;
                const int t = t0 + tt * 16 + ln;
                c[r] = c[r] * 0.125f + mask[(size_t)q * SEQ + t];
            }
            s[tt] = c;
        }

        // ---- online softmax row update (rows live across a 16-lane group) ----
#pragma unroll
        for (int r = 0; r < 8; ++r) {
            float tmax = fmaxf(s[0][r], s[1][r]);
#pragma unroll
            for (int off = 1; off < 16; off <<= 1)
                tmax = fmaxf(tmax, __shfl_xor(tmax, off, 16));
            const float mn = fmaxf(m_r[r], tmax);
            const float al = __expf(m_r[r] - mn);
            m_r[r] = mn;
            const float p0 = __expf(s[0][r] - mn);
            const float p1 = __expf(s[1][r] - mn);
            float rs = p0 + p1;
#pragma unroll
            for (int off = 1; off < 16; off <<= 1)
                rs += __shfl_xor(rs, off, 16);
            l_r[r] = l_r[r] * al + rs;
            s[0][r] = p0;
            s[1][r] = p1;
#pragma unroll
            for (int j = 0; j < 4; ++j) o[j][r] *= al;
        }

        // ---- re-layout P (C-frag -> A-frag) through per-wave LDS ----
        bf16* pl = Plds + wave * 512;
#pragma unroll
        for (int tt = 0; tt < 2; ++tt)
#pragma unroll
            for (int r = 0; r < 8; ++r)
                pl[(r + 8 * half) * 32 + tt * 16 + ln] = f2bf(s[tt][r]);
        // DS ops are in-order within a wave: safe to read back without barrier
        bf16x16 pa = loadA(pl + ln * 32, half);

        // ---- O += P x V  (V transposed so B-operand is contiguous along t) ----
#pragma unroll
        for (int j = 0; j < 4; ++j) {
            const bf16* vrow = Vtbase + (size_t)(j * 16 + ln) * SEQ + t0;
            bf16x16 bv = load16(vrow + half * 16);
            o[j] = wmma_bf16(pa, bv, o[j]);
        }
    }

    // ---- normalize and write [B,S,H*Dh] bf16 ----
#pragma unroll
    for (int j = 0; j < 4; ++j) {
#pragma unroll
        for (int r = 0; r < 8; ++r) {
            const int q = q0 + r + 8 * half;
            const float v = o[j][r] / l_r[r];
            attnb[((size_t)b * SEQ + q) * EMBED + h * HDIM + j * 16 + ln] = f2bf(v);
        }
    }
}

// ---------- launch ----------

extern "C" void kernel_launch(void* const* d_in, const int* in_sizes, int n_in,
                              void* d_out, int out_size, void* d_ws, size_t ws_size,
                              hipStream_t stream) {
    const float* x    = (const float*)d_in[0];
    const float* mask = (const float*)d_in[1];
    const float* Wq = (const float*)d_in[2]; const float* bq = (const float*)d_in[3];
    const float* Wk = (const float*)d_in[4]; const float* bk = (const float*)d_in[5];
    const float* Wv = (const float*)d_in[6]; const float* bv = (const float*)d_in[7];
    const float* Wo = (const float*)d_in[8]; const float* bo = (const float*)d_in[9];

    char* ws = (char*)d_ws;
    const size_t MB = 1024 * 1024;
    bf16* xb  = (bf16*)(ws);                    //  8 MB : x bf16 [4096,1024]
    bf16* Wqb = (bf16*)(ws + 8 * MB);           //  2 MB each
    bf16* Wkb = Wqb + EMBED * EMBED;
    bf16* Wvb = Wkb + EMBED * EMBED;
    bf16* Wob = Wvb + EMBED * EMBED;
    bf16* Qb  = (bf16*)(ws + 16 * MB);          //  8 MB : [B,H,S,Dh]
    bf16* Kb  = Qb + (size_t)MTOT * EMBED;      //  8 MB : [B,H,S,Dh]
    bf16* Vtb = Kb + (size_t)MTOT * EMBED;      //  8 MB : [B,H,Dh,S]
    bf16* Ab  = Vtb + (size_t)MTOT * EMBED;     //  8 MB : [B,S,E]  (attn out)

    cvt_f32_bf16<<<2048, 256, 0, stream>>>(x, xb, MTOT * EMBED);
    cvt_f32_bf16<<<512, 256, 0, stream>>>(Wq, Wqb, EMBED * EMBED);
    cvt_f32_bf16<<<512, 256, 0, stream>>>(Wk, Wkb, EMBED * EMBED);
    cvt_f32_bf16<<<512, 256, 0, stream>>>(Wv, Wvb, EMBED * EMBED);
    cvt_f32_bf16<<<512, 256, 0, stream>>>(Wo, Wob, EMBED * EMBED);

    dim3 gg(EMBED / 128, MTOT / 128);   // (8, 32) blocks, 128 threads (4 waves, 2x2)
    gemm_bf16_wmma<<<gg, 128, 0, stream>>>(xb, Wqb, bq, Qb, 0);
    gemm_bf16_wmma<<<gg, 128, 0, stream>>>(xb, Wkb, bk, Kb, 0);
    gemm_bf16_wmma<<<gg, 128, 0, stream>>>(xb, Wvb, bv, Vtb, 1);

    attn_wmma<<<BATCH * HEADS * (SEQ / 16) / 4, 128, 0, stream>>>(Qb, Kb, Vtb, mask, Ab);

    gemm_bf16_wmma<<<gg, 128, 0, stream>>>(Ab, Wob, bo, d_out, 2);
}
